// FMGRUModel_85109071937746
// MI455X (gfx1250) — matrix operand order repaired and checked by reading, not compiled, not run on hardware
//
#include <hip/hip_runtime.h>
#include <hip/hip_bf16.h>
#include <math.h>

// ---------------------------------------------------------------------------
// FM-GRU forward for MI455X (gfx1250, wave32, WMMA).
// Heavy GEMMs: v_wmma_f32_16x16x32_f16 with double-buffered LDS tile staging
// via the CDNA5 async global->LDS path (ASYNCcnt + s_wait_asynccnt), falling
// back to a VGPR-bounce copy if the async builtins are unavailable.
// ---------------------------------------------------------------------------

typedef _Float16 f16;
typedef __attribute__((ext_vector_type(16))) _Float16 v16h;
typedef __attribute__((ext_vector_type(8)))  float    v8f;
typedef int v4i __attribute__((vector_size(16)));  // matches builtin prototype

union HFrag { v16h v; uint4 u[2]; };

#if defined(__HIP_DEVICE_COMPILE__) && \
    __has_builtin(__builtin_amdgcn_global_load_async_to_lds_b128)
#define USE_ASYNC_LDS 1
#else
#define USE_ASYNC_LDS 0
#endif

__device__ __forceinline__ void wait_async0() {
#if USE_ASYNC_LDS
#if __has_builtin(__builtin_amdgcn_s_wait_asynccnt)
  __builtin_amdgcn_s_wait_asynccnt(0);
#else
  asm volatile("s_wait_asynccnt 0" ::: "memory");
#endif
#endif
}

__device__ __forceinline__ float wave_sum(float x) {
#pragma unroll
  for (int o = 16; o > 0; o >>= 1) x += __shfl_xor(x, o);
  return x;
}
__device__ __forceinline__ float wave_max(float x) {
#pragma unroll
  for (int o = 16; o > 0; o >>= 1) x = fmaxf(x, __shfl_xor(x, o));
  return x;
}

// LDS tile geometry: 64 A rows + 256 B rows, 32 K-halves per row,
// row stride padded to 40 halves (80 B) -> fragment reads are bank-conflict
// free (16 lanes land on 16 distinct 4-bank groups of the 64 LDS banks).
#define SROW 40
#define TROWS 320  // 64 (A) + 256 (B)

// Copy one 64x32 A tile + 256x32 B tile into LDS buffer `sbuf`.
// 1280 16-byte chunks / 256 threads = 5 async b128 copies per thread.
__device__ __forceinline__ void issue_tile_copy(
    const f16* __restrict__ A, int lda, int m0blk,
    const f16* __restrict__ W, int ldw, int n0blk,
    int k0, f16* __restrict__ sbuf, int tid) {
#pragma unroll
  for (int j = 0; j < 5; ++j) {
    const int c = tid + 256 * j;  // 0..1279
    const int r = c >> 2, p = c & 3;
    const f16* src = (r < 64)
        ? (A + (size_t)(m0blk + r) * lda + k0 + p * 8)
        : (W + (size_t)(n0blk + (r - 64)) * ldw + k0 + p * 8);
    f16* dst = sbuf + r * SROW + p * 8;
#if USE_ASYNC_LDS
    __builtin_amdgcn_global_load_async_to_lds_b128((v4i*)src, (v4i*)dst, 0, 0);
#else
    *(uint4*)dst = *(const uint4*)src;
#endif
  }
}

// C[M,N] = A[M,K] @ W[N,K]^T + bias, f16 operands, f32 accumulate.
// Block: 256 thr = 8 waves (2x4); wave tile 32x64 -> block tile 64x256.
// M multiple of 64 (all call sites); K multiple of 32 (buffers zero-padded);
// N guarded on store (weight rows padded to the 256-col block).
template <typename OutT>
__global__ __launch_bounds__(256) void wmma_gemm_kern(
    const f16* __restrict__ A, int lda, const f16* __restrict__ W, int ldw,
    OutT* __restrict__ C, int ldc, int N, int K, const float* __restrict__ bias)
{
  __shared__ f16 smem[2][TROWS * SROW];  // 2 x 25600 B
  const int tid  = threadIdx.x;
  const int lane = tid & 31;
  const int wave = tid >> 5;
  const int wm = wave >> 2, wn = wave & 3;
  const int m0blk = blockIdx.y * 64;
  const int n0blk = blockIdx.x * 256;
  const int ml = lane & 15;
  const int g  = lane >> 4;

  v8f acc[2][4];
#pragma unroll
  for (int mi = 0; mi < 2; ++mi)
#pragma unroll
    for (int ni = 0; ni < 4; ++ni)
#pragma unroll
      for (int r = 0; r < 8; ++r) acc[mi][ni][r] = 0.f;

  const int nk = K >> 5;
  issue_tile_copy(A, lda, m0blk, W, ldw, n0blk, 0, smem[0], tid);
  wait_async0();
  __syncthreads();

  for (int ki = 0; ki < nk; ++ki) {
    const int cur = ki & 1;
    if (ki + 1 < nk)  // prefetch next K tile into the other buffer
      issue_tile_copy(A, lda, m0blk, W, ldw, n0blk, (ki + 1) * 32,
                      smem[cur ^ 1], tid);

    const f16* sA = &smem[cur][0];
    const f16* sB = &smem[cur][64 * SROW];
    HFrag a[2], b[4];
#pragma unroll
    for (int mi = 0; mi < 2; ++mi) {
      // A 16-bit layout (ISA 7.12.2): halves i -> K = 8g + (i<8 ? i : i+8)
      const f16* pa = sA + (wm * 32 + mi * 16 + ml) * SROW + 8 * g;
      a[mi].u[0] = *(const uint4*)pa;
      a[mi].u[1] = *(const uint4*)(pa + 16);
    }
#pragma unroll
    for (int ni = 0; ni < 4; ++ni) {
      // B 16-bit layout: halves i -> K = 8g + (i<8 ? i : i+8)
      const f16* pb = sB + (wn * 64 + ni * 16 + ml) * SROW + 8 * g;
      b[ni].u[0] = *(const uint4*)pb;
      b[ni].u[1] = *(const uint4*)(pb + 16);
    }
#pragma unroll
    for (int mi = 0; mi < 2; ++mi)
#pragma unroll
      for (int ni = 0; ni < 4; ++ni)
        acc[mi][ni] = __builtin_amdgcn_wmma_f32_16x16x32_f16(
            false, a[mi].v, false, b[ni].v, (short)0, acc[mi][ni], false, false);

    if (ki + 1 < nk) {
      wait_async0();     // next tile landed in LDS
      __syncthreads();   // and everyone is done reading the current one
    }
  }

#pragma unroll
  for (int ni = 0; ni < 4; ++ni) {
    const int n = n0blk + wn * 64 + ni * 16 + ml;
    if (n >= N) continue;
    const float bv = bias ? bias[n] : 0.f;
#pragma unroll
    for (int mi = 0; mi < 2; ++mi)
#pragma unroll
      for (int r = 0; r < 8; ++r) {
        const int m = m0blk + wm * 32 + mi * 16 + r + 8 * g;
        C[(size_t)m * ldc + n] = (OutT)(acc[mi][ni][r] + bv);
      }
  }
}

// ------------------------- small prep / elementwise -------------------------

__global__ void k_cvt_pad(const float* __restrict__ s, f16* __restrict__ d,
                          int rows, int scols, int dcols) {
  long i = (long)blockIdx.x * blockDim.x + threadIdx.x;
  if (i >= (long)rows * dcols) return;
  int r = (int)(i / dcols), c = (int)(i % dcols);
  d[i] = (c < scols) ? (f16)s[(size_t)r * scols + c] : (f16)0.f;
}

__global__ void k_fmvT(const float* __restrict__ fmv, f16* __restrict__ d) {
  int i = blockIdx.x * blockDim.x + threadIdx.x;         // 256*512
  if (i >= 256 * 512) return;
  int j = i >> 9, k = i & 511;
  d[i] = (j < 84) ? (f16)fmv[(size_t)k * 84 + j] : (f16)0.f;
}

__global__ void k_histxT(const float* __restrict__ hx, f16* __restrict__ d) {
  long i = (long)blockIdx.x * blockDim.x + threadIdx.x;  // 49152*32
  if (i >= 49152L * 32) return;
  int row = (int)(i >> 5), f = (int)(i & 31);
  int t = row >> 11, b = row & 2047;                      // row = t*2048+b
  d[i] = (f16)hx[((size_t)b * 24 + t) * 32 + f];
}

__global__ void k_futxT(const float* __restrict__ fx, f16* __restrict__ d) {
  long i = (long)blockIdx.x * blockDim.x + threadIdx.x;  // 24576*32
  if (i >= 24576L * 32) return;
  int row = (int)(i >> 5), f = (int)(i & 31);
  int t = row >> 11, b = row & 2047;
  d[i] = (f16)fx[((size_t)b * 12 + t) * 32 + f];
}

__global__ void k_vs(const float* __restrict__ fmv, float* __restrict__ vs) {
  int k = blockIdx.x * blockDim.x + threadIdx.x;
  if (k >= 512) return;
  float s = 0.f;
  for (int j = 0; j < 84; ++j) { float v = fmv[(size_t)k * 84 + j]; s += v * v; }
  vs[k] = s;
}

__global__ void k_bq2(const float* __restrict__ bq, const float* __restrict__ bt,
                      float* __restrict__ d) {
  int i = blockIdx.x * blockDim.x + threadIdx.x;
  if (i >= 1024) return;
  d[i] = (i < 512) ? bq[i] : bt[i - 512];
}

// z_emb: enc_in cols 0..511 = histz[b,t]*W_ie[n] + b_ie[n]   (rows t*2048+b)
__global__ void k_zemb(const float* __restrict__ histz, const float* __restrict__ Wie,
                       const float* __restrict__ bie, f16* __restrict__ enc_in) {
  long i = (long)blockIdx.x * blockDim.x + threadIdx.x;  // 49152*512
  if (i >= 49152L * 512) return;
  int row = (int)(i >> 9), n = (int)(i & 511);
  int t = row >> 11, b = row & 2047;
  float v = histz[(size_t)b * 24 + t] * Wie[n] + bie[n];
  enc_in[(size_t)row * 1056 + n] = (f16)v;
}

// FM term: fm = x.fm_w + fm_b + 0.5*(x^2 . vs - sum_j p1_j^2); zero-pads tail.
__global__ void k_fm(f16* __restrict__ inb, int ld, int xoff, int fmcol,
                     const float* __restrict__ P1, const float* __restrict__ fm_w,
                     const float* __restrict__ fm_b, const float* __restrict__ vs,
                     int rows) {
  int r = blockIdx.x * blockDim.x + threadIdx.x;
  if (r >= rows) return;
  const f16* x = inb + (size_t)r * ld + xoff;
  float lin = fm_b[0], q2 = 0.f;
  for (int k = 0; k < 512; ++k) {
    float xv = (float)x[k];
    lin += xv * fm_w[k];
    q2  += xv * xv * vs[k];
  }
  float s = 0.f;
  for (int j = 0; j < 96; ++j) { float p = P1[(size_t)r * 96 + j]; s += p * p; }
  float fmv = lin + 0.5f * (q2 - s);
  f16* row = inb + (size_t)r * ld;
  row[fmcol] = (f16)fmv;
  for (int c = fmcol + 1; c < ld; ++c) row[c] = (f16)0.f;
}

// GRU gate fusion: h' = (1-z)*tanh(gi_n + r*gh_n) + z*h
__global__ void k_gate(const float* __restrict__ Gi, const float* __restrict__ Gh,
                       float* __restrict__ h, f16* __restrict__ h16,
                       float* __restrict__ st, int BH) {
  int i = blockIdx.x * blockDim.x + threadIdx.x;
  if (i >= BH) return;
  int b = i >> 9, n = i & 511;
  size_t base = (size_t)b * 1536;
  float r  = 1.f / (1.f + expf(-(Gi[base + n]        + Gh[base + n])));
  float z  = 1.f / (1.f + expf(-(Gi[base + 512 + n]  + Gh[base + 512 + n])));
  float nn = tanhf(Gi[base + 1024 + n] + r * Gh[base + 1024 + n]);
  float hn = (1.f - z) * nn + z * h[i];
  h[i] = hn;
  h16[i] = (f16)hn;
  st[i] = hn;
}

// Per-decoder-step attention + TPA; one block per batch element b.
__global__ __launch_bounds__(256) void k_attn(
    const f16* __restrict__ kenc, const f16* __restrict__ qqt,
    const float* __restrict__ tpaWk, const float* __restrict__ states,
    const float* __restrict__ futx, f16* __restrict__ dec_in_t) {
  __shared__ float s_sc[24], s_m[12], s_at[32];
  const int b = blockIdx.x;
  const int lane = threadIdx.x & 31, w = threadIdx.x >> 5;

  // encoder attention scores: wave w does t = w, w+8, w+16
  for (int t = w; t < 24; t += 8) {
    const f16* kr = kenc + ((size_t)t * 2048 + b) * 512;
    const f16* qr = qqt + (size_t)b * 1024;
    float p = 0.f;
    for (int h = lane; h < 512; h += 32) p += (float)kr[h] * (float)qr[h];
    p = wave_sum(p);
    if (lane == 0) s_sc[t] = p;
  }
  // m[t] = sum_h tpa_Wk[h,t] * qt[h]  (TPA score factorization; key bias is
  // softmax-invariant and dropped)
  for (int t = w; t < 12; t += 8) {
    const f16* qr = qqt + (size_t)b * 1024 + 512;
    float p = 0.f;
    for (int h = lane; h < 512; h += 32) p += tpaWk[(size_t)h * 12 + t] * (float)qr[h];
    p = wave_sum(p);
    if (lane == 0) s_m[t] = p;
  }
  __syncthreads();

  if (w == 0) {  // softmax over 24 encoder scores
    float v = (lane < 24) ? s_sc[lane] : -1e30f;
    float mx = wave_max(v);
    float e = (lane < 24) ? expf(v - mx) : 0.f;
    float sm = wave_sum(e);
    if (lane < 24) s_sc[lane] = e / sm;
  }
  if (w == 1) {  // TPA scores over 32 features + softmax
    float sc = 0.f;
    for (int t = 0; t < 12; ++t) sc += futx[((size_t)b * 12 + t) * 32 + lane] * s_m[t];
    float mx = wave_max(sc);
    float e = expf(sc - mx);
    float sm = wave_sum(e);
    s_at[lane] = e / sm;
  }
  __syncthreads();

  f16* dst = dec_in_t + (size_t)b * 1056;
  for (int n = threadIdx.x; n < 512; n += 256) {  // attn_out -> cols 512..1023
    float a = 0.f;
    for (int t = 0; t < 24; ++t)
      a += s_sc[t] * states[((size_t)t * 2048 + b) * 512 + n];
    dst[512 + n] = (f16)a;
  }
  if (threadIdx.x < 12) {                          // tpa_out -> cols 1024..1035
    float o = 0.f;
    for (int f = 0; f < 32; ++f)
      o += futx[((size_t)b * 12 + threadIdx.x) * 32 + f] * s_at[f];
    dst[1024 + threadIdx.x] = (f16)o;
  }
}

// Output heads: preds/future/mu/sigma from all 36 states.
__global__ void k_out(const float* __restrict__ states,
                      const float* __restrict__ oW, const float* __restrict__ ob,
                      const float* __restrict__ mW, const float* __restrict__ mb,
                      const float* __restrict__ gW, const float* __restrict__ gb,
                      float* __restrict__ out) {
  int i = blockIdx.x * blockDim.x + threadIdx.x;  // 73728 = 2048*36
  if (i >= 73728) return;
  int b = i / 36, s = i % 36;
  const float* sp = states + ((size_t)s * 2048 + b) * 512;
  float p = ob[0], m = mb[0], g = gb[0];
  for (int k = 0; k < 512; ++k) {
    float v = sp[k];
    p += fmaxf(v, 0.f) * oW[k];
    m += v * mW[k];
    g += v * gW[k];
  }
  out[i] = p;
  if (s >= 24) out[73728 + (size_t)b * 12 + (s - 24)] = p;
  out[98304 + i] = m;
  out[172032 + i] = g;
}

// ------------------------------- host driver -------------------------------

extern "C" void kernel_launch(void* const* d_in, const int* in_sizes, int n_in,
                              void* d_out, int out_size, void* d_ws, size_t ws_size,
                              hipStream_t stream) {
  (void)in_sizes; (void)n_in; (void)out_size;
  const int B = 2048, TE = 24, TF = 12, H = 512, H3 = 1536, KP = 1056;

  const float* histx   = (const float*)d_in[0];
  const float* histz   = (const float*)d_in[1];
  const float* futx    = (const float*)d_in[2];
  const float* W_ie    = (const float*)d_in[4];
  const float* b_ie    = (const float*)d_in[5];
  const float* W_fe    = (const float*)d_in[6];
  const float* b_fe    = (const float*)d_in[7];
  const float* fm_w    = (const float*)d_in[8];
  const float* fm_b    = (const float*)d_in[9];
  const float* fm_v    = (const float*)d_in[10];
  const float* enc_Wih = (const float*)d_in[11];
  const float* enc_Whh = (const float*)d_in[12];
  const float* enc_bih = (const float*)d_in[13];
  const float* enc_bhh = (const float*)d_in[14];
  const float* dec_Wih = (const float*)d_in[15];
  const float* dec_Whh = (const float*)d_in[16];
  const float* dec_bih = (const float*)d_in[17];
  const float* dec_bhh = (const float*)d_in[18];
  const float* attn_Wq = (const float*)d_in[19];
  const float* attn_bq = (const float*)d_in[20];
  const float* attn_Wk = (const float*)d_in[21];
  const float* attn_bk = (const float*)d_in[22];
  const float* tpa_Wq  = (const float*)d_in[23];
  const float* tpa_bq  = (const float*)d_in[24];
  const float* tpa_Wk  = (const float*)d_in[25];
  const float* out_W   = (const float*)d_in[27];
  const float* out_b   = (const float*)d_in[28];
  const float* mu_W    = (const float*)d_in[29];
  const float* mu_b    = (const float*)d_in[30];
  const float* sig_W   = (const float*)d_in[31];
  const float* sig_b   = (const float*)d_in[32];
  float* out = (float*)d_out;

  // ---- workspace carve (bump allocator, 256B aligned) ----
  char* cur = (char*)d_ws;
  auto alloc = [&](size_t bytes) -> void* {
    void* r = (void*)cur;
    cur += (bytes + 255) & ~(size_t)255;
    return r;
  };
  f16*   wih_e  = (f16*)alloc((size_t)H3 * KP * 2);
  f16*   whh_e  = (f16*)alloc((size_t)H3 * H * 2);
  f16*   wih_d  = (f16*)alloc((size_t)H3 * KP * 2);
  f16*   whh_d  = (f16*)alloc((size_t)H3 * H * 2);
  f16*   wk16   = (f16*)alloc((size_t)H * H * 2);
  f16*   wq2    = (f16*)alloc((size_t)1024 * H * 2);
  f16*   wfe16  = (f16*)alloc((size_t)H * 32 * 2);
  f16*   fmvT   = (f16*)alloc((size_t)256 * H * 2);
  f16*   histxT = (f16*)alloc((size_t)TE * B * 32 * 2);
  f16*   futxT  = (f16*)alloc((size_t)TF * B * 32 * 2);
  float* vs     = (float*)alloc(512 * 4);
  float* bq2    = (float*)alloc(1024 * 4);
  f16*   enc_in = (f16*)alloc((size_t)TE * B * KP * 2);
  f16*   dec_in = (f16*)alloc((size_t)TF * B * KP * 2);
  float* P1     = (float*)alloc((size_t)TE * B * 96 * 4);
  float* P1d    = (float*)alloc((size_t)TF * B * 96 * 4);
  float* Gi     = (float*)alloc((size_t)B * H3 * 4);
  float* Gh     = (float*)alloc((size_t)B * H3 * 4);
  float* hf     = (float*)alloc((size_t)B * H * 4);
  f16*   h16    = (f16*)alloc((size_t)B * H * 2);
  f16*   kenc   = (f16*)alloc((size_t)TE * B * H * 2);
  f16*   qqt    = (f16*)alloc((size_t)B * 1024 * 2);
  float* states = (float*)alloc((size_t)(TE + TF) * B * H * 4);
  if ((size_t)(cur - (char*)d_ws) > ws_size) return;  // ws too small: bail

  auto gsz = [](long n) { return dim3((unsigned)((n + 255) / 256)); };
  auto gemmF = [&](const f16* A, int lda, const f16* W, int ldw, float* C, int ldc,
                   int M, int N, int K, const float* bias) {
    dim3 grid((N + 255) / 256, M / 64);
    wmma_gemm_kern<float><<<grid, 256, 0, stream>>>(A, lda, W, ldw, C, ldc, N, K, bias);
  };
  auto gemmH = [&](const f16* A, int lda, const f16* W, int ldw, f16* C, int ldc,
                   int M, int N, int K, const float* bias) {
    dim3 grid((N + 255) / 256, M / 64);
    wmma_gemm_kern<f16><<<grid, 256, 0, stream>>>(A, lda, W, ldw, C, ldc, N, K, bias);
  };

  // ---- prep: f16 weight conversion (+K pad), transposed operand builds ----
  k_cvt_pad<<<gsz((long)H3 * KP), 256, 0, stream>>>(enc_Wih, wih_e, H3, 1025, KP);
  k_cvt_pad<<<gsz((long)H3 * H), 256, 0, stream>>>(enc_Whh, whh_e, H3, H, H);
  k_cvt_pad<<<gsz((long)H3 * KP), 256, 0, stream>>>(dec_Wih, wih_d, H3, 1037, KP);
  k_cvt_pad<<<gsz((long)H3 * H), 256, 0, stream>>>(dec_Whh, whh_d, H3, H, H);
  k_cvt_pad<<<gsz((long)H * H), 256, 0, stream>>>(attn_Wk, wk16, H, H, H);
  k_cvt_pad<<<gsz((long)H * H), 256, 0, stream>>>(attn_Wq, wq2, H, H, H);
  k_cvt_pad<<<gsz((long)H * H), 256, 0, stream>>>(tpa_Wq, wq2 + (size_t)H * H, H, H, H);
  k_cvt_pad<<<gsz((long)H * 32), 256, 0, stream>>>(W_fe, wfe16, H, 32, 32);
  k_fmvT<<<gsz(256L * H), 256, 0, stream>>>(fm_v, fmvT);
  k_histxT<<<gsz((long)TE * B * 32), 256, 0, stream>>>(histx, histxT);
  k_futxT<<<gsz((long)TF * B * 32), 256, 0, stream>>>(futx, futxT);
  k_vs<<<gsz(512), 256, 0, stream>>>(fm_v, vs);
  k_bq2<<<gsz(1024), 256, 0, stream>>>(attn_bq, tpa_bq, bq2);
  (void)hipMemsetAsync(hf, 0, (size_t)B * H * 4, stream);
  (void)hipMemsetAsync(h16, 0, (size_t)B * H * 2, stream);

  // ---- encoder inputs: z_emb | x_emb | fm  ->  enc_in [TE*B, 1056] f16 ----
  k_zemb<<<gsz((long)TE * B * H), 256, 0, stream>>>(histz, W_ie, b_ie, enc_in);
  gemmH(histxT, 32, wfe16, 32, enc_in + 512, KP, TE * B, H, 32, b_fe);     // x_emb
  gemmF(enc_in + 512, KP, fmvT, H, P1, 96, TE * B, 96, H, nullptr);        // x@fm_v
  k_fm<<<gsz(TE * B), 256, 0, stream>>>(enc_in, KP, 512, 1024, P1, fm_w, fm_b, vs, TE * B);

  // ---- encoder GRU scan (per-step WMMA GEMMs + fused gates) ----
  for (int t = 0; t < TE; ++t) {
    gemmF(enc_in + (size_t)t * B * KP, KP, wih_e, KP, Gi, H3, B, H3, KP, enc_bih);
    gemmF(h16, H, whh_e, H, Gh, H3, B, H3, H, enc_bhh);
    k_gate<<<gsz((long)B * H), 256, 0, stream>>>(Gi, Gh, hf, h16,
                                                 states + (size_t)t * B * H, B * H);
    gemmH(h16, H, wk16, H, kenc + (size_t)t * B * H, H, B, H, H, attn_bk); // k_enc[t]
  }

  // ---- decoder precompute: f_emb | fm -> dec_in skeleton ----
  gemmH(futxT, 32, wfe16, 32, dec_in, KP, TF * B, H, 32, b_fe);            // f_emb
  gemmF(dec_in, KP, fmvT, H, P1d, 96, TF * B, 96, H, nullptr);
  k_fm<<<gsz(TF * B), 256, 0, stream>>>(dec_in, KP, 0, 1036, P1d, fm_w, fm_b, vs, TF * B);

  // ---- decoder scan ----
  for (int t = 0; t < TF; ++t) {
    gemmH(h16, H, wq2, H, qqt, 1024, B, 1024, H, bq2);                     // q | qt
    k_attn<<<dim3(B), 256, 0, stream>>>(kenc, qqt, tpa_Wk, states, futx,
                                        dec_in + (size_t)t * B * KP);
    gemmF(dec_in + (size_t)t * B * KP, KP, wih_d, KP, Gi, H3, B, H3, KP, dec_bih);
    gemmF(h16, H, whh_d, H, Gh, H3, B, H3, H, dec_bhh);
    k_gate<<<gsz((long)B * H), 256, 0, stream>>>(Gi, Gh, hf, h16,
                                                 states + (size_t)(TE + t) * B * H, B * H);
  }

  // ---- output heads ----
  k_out<<<gsz(73728), 256, 0, stream>>>(states, out_W, out_b, mu_W, mu_b,
                                        sig_W, sig_b, out);
}
